// DiffusionLayer_rec2_transformer_78426102825600
// MI455X (gfx1250) — matrix operand order repaired and checked by reading, not compiled, not run on hardware
//
#include <hip/hip_runtime.h>
#include <cstdint>
#include <cstddef>

// Point-transformer layer for MI455X (gfx1250), wave32 + WMMA bf16.
// Workspace use: ~190 MB (idx 1MB, six [B,128,N] f32 bufs 8MB each,
// pos1/att1 bf16 67MB each, small stats). B=2, N=8192, FEAT=128, K=16.

#define FEAT 128
#define GROUPS 8
#define KNN 16
#define NEG_SLOPE 0.1f
#define GEPS 1e-5f
#define XS 132  // LDS X column stride in bf16 elems (bank-conflict padding)

typedef __bf16 bf16;
typedef bf16  v16bf __attribute__((ext_vector_type(16)));
typedef float v8f   __attribute__((ext_vector_type(8)));

__device__ __forceinline__ bf16 f2bf(float x) { return (bf16)x; }
__device__ __forceinline__ float bf2f(bf16 x) { return (float)x; }
__device__ __forceinline__ float leaky(float x) { return x >= 0.f ? x : NEG_SLOPE * x; }

// ---- WMMA fragment loads (per CDNA5 ISA 7.12.2 layouts) ----
// A 16x32 bf16: lane L holds row M=L%16; elems 0..7 -> K = kt*32 + 8*(L/16) + e,
// elems 8..15 -> K = kt*32 + 16 + 8*(L/16) + (e-8). W in LDS row-major [128][128].
__device__ __forceinline__ v16bf load_a_frag(const bf16* Wl, int mt, int kt, int l16, int half) {
  const bf16* pa = &Wl[(mt * 16 + l16) * FEAT + kt * 32 + 8 * half];
  v16bf a;
#pragma unroll
  for (int i = 0; i < 8; ++i) { a[i] = pa[i]; a[i + 8] = pa[i + 16]; }
  return a;
}
// B 32x16 bf16: lane L holds col N=L%16; elems 0..7 -> K = kt*32 + 8*(L/16) + e,
// elems 8..15 -> K = kt*32 + 16 + 8*(L/16) + (e-8).
// X in LDS column-major: Xl[col*XS + ch].
__device__ __forceinline__ v16bf load_b_frag(const bf16* Xl, int jt, int kt, int l16, int half) {
  const bf16* pb = &Xl[(jt + l16) * XS + kt * 32 + 8 * half];
  v16bf b;
#pragma unroll
  for (int i = 0; i < 8; ++i) { b[i] = pb[i]; b[i + 8] = pb[i + 16]; }
  return b;
}

// =====================  KNN  =====================
// One thread per query point; tiles of 256 candidate points staged in LDS.
__global__ __launch_bounds__(256) void knn_kernel(const float* __restrict__ xyz,
                                                  int* __restrict__ idx, int N) {
  const int p = blockIdx.x * 256 + threadIdx.x;   // global point in [0, B*N)
  const int b = p / N, n = p - b * N;
  const float* base = xyz + (size_t)b * 3 * N;
  const float qx = base[n], qy = base[N + n], qz = base[2 * N + n];
  float bd[KNN]; int bi[KNN];
#pragma unroll
  for (int i = 0; i < KNN; ++i) { bd[i] = 3.0e38f; bi[i] = 0; }
  float worst = 3.0e38f; int wp = KNN - 1;
  __shared__ float px[256], py[256], pz[256];
  for (int t0 = 0; t0 < N; t0 += 256) {
    __syncthreads();
    const int j = t0 + threadIdx.x;
    px[threadIdx.x] = base[j]; py[threadIdx.x] = base[N + j]; pz[threadIdx.x] = base[2 * N + j];
    __syncthreads();
    for (int t = 0; t < 256; ++t) {
      const float dx = px[t] - qx, dy = py[t] - qy, dz = pz[t] - qz;
      const float d = dx * dx + dy * dy + dz * dz;
      if (d < worst) {
        bd[wp] = d; bi[wp] = t0 + t;
        worst = bd[0]; wp = 0;
#pragma unroll
        for (int i = 1; i < KNN; ++i) if (bd[i] > worst) { worst = bd[i]; wp = i; }
      }
    }
  }
  int* op = idx + (size_t)p * KNN;
#pragma unroll
  for (int i = 0; i < KNN; ++i) op[i] = bi[i];
}

// =====================  Generic 128xK GEMM: Y = W @ X + bias  =====================
// X,Y: [B,128,N] f32 (cols = B*N flattened, N-major). 128 threads = 4 waves,
// each wave one 16-col tile, 64 cols/block. Optional GroupNorm stat accumulation.
__global__ __launch_bounds__(128) void gemm128_kernel(
    const float* __restrict__ W, const float* __restrict__ X,
    const float* __restrict__ bias, float* __restrict__ Y,
    int N, int cols, double* __restrict__ stats) {
  __shared__ bf16 Wl[FEAT * FEAT];
  __shared__ bf16 Xl[64 * XS];
  __shared__ float biasl[FEAT];
  const int tid = threadIdx.x;
  for (int i = tid; i < FEAT * FEAT; i += 128) Wl[i] = f2bf(W[i]);
  if (tid < FEAT) biasl[tid] = bias ? bias[tid] : 0.0f;
  const int c0 = blockIdx.x * 64;
  for (int e = tid; e < 64 * FEAT; e += 128) {
    const int j = e & 63, ch = e >> 6;
    const int c = c0 + j;
    float v = 0.0f;
    if (c < cols) { const int b = c / N, n = c - b * N; v = X[((size_t)b * FEAT + ch) * N + n]; }
    Xl[j * XS + ch] = f2bf(v);
  }
  __syncthreads();
  const int wave = tid >> 5, lane = tid & 31, half = lane >> 4, l16 = lane & 15;
  const int c = c0 + wave * 16 + l16;
  const int cc = c < cols ? c : cols - 1;
  const int b = cc / N, n = cc - b * N;
  for (int mt = 0; mt < 8; ++mt) {
    v8f acc;
#pragma unroll
    for (int v = 0; v < 8; ++v) acc[v] = biasl[mt * 16 + v + 8 * half];
#pragma unroll
    for (int kt = 0; kt < 4; ++kt) {
      v16bf a = load_a_frag(Wl, mt, kt, l16, half);
      v16bf bm = load_b_frag(Xl, wave * 16, kt, l16, half);
      acc = __builtin_amdgcn_wmma_f32_16x16x32_bf16(false, a, false, bm, (short)0, acc, false, false);
    }
#pragma unroll
    for (int v = 0; v < 8; ++v) {
      const int ch = mt * 16 + v + 8 * half;
      if (c < cols) Y[((size_t)b * FEAT + ch) * N + n] = acc[v];
    }
    if (stats) {
      double s = 0.0, q = 0.0;
#pragma unroll
      for (int v = 0; v < 8; ++v) { const double d = (c < cols) ? (double)acc[v] : 0.0; s += d; q += d * d; }
#pragma unroll
      for (int off = 16; off >= 1; off >>= 1) { s += __shfl_xor(s, off, 32); q += __shfl_xor(q, off, 32); }
      if (lane == 0) {
        const int bB = c0 / N;   // whole block in one batch (N % 64 == 0)
        atomicAdd(&stats[(bB * GROUPS + mt) * 2 + 0], s);
        atomicAdd(&stats[(bB * GROUPS + mt) * 2 + 1], q);
      }
    }
  }
}

// =====================  pos1 = W_pos1 @ rel + b_pos1  =====================
// Thread per (b,ch,n); writes 16 neighbor values, accumulates GroupNorm stats.
__global__ __launch_bounds__(256) void pos1_kernel(
    const float* __restrict__ xyz, const int* __restrict__ idx,
    const float* __restrict__ Wp, const float* __restrict__ bp,
    bf16* __restrict__ pos1, double* __restrict__ stats, int N) {
  const int t = blockIdx.x * 256 + threadIdx.x;
  const int b = t / (FEAT * N);
  const int rem = t - b * FEAT * N;
  const int ch = rem / N, n = rem - ch * N;
  const float w0 = Wp[ch * 3 + 0], w1 = Wp[ch * 3 + 1], w2 = Wp[ch * 3 + 2], bb = bp[ch];
  const float* xb = xyz + (size_t)b * 3 * N;
  const float cx = xb[n], cy = xb[N + n], cz = xb[2 * N + n];
  const int* ip = idx + ((size_t)b * N + n) * KNN;
  bf16* op = pos1 + (((size_t)b * FEAT + ch) * N + n) * KNN;
  double s = 0.0, q = 0.0;
#pragma unroll
  for (int k = 0; k < KNN; ++k) {
    const int j = ip[k];
    const float val = w0 * (xb[j] - cx) + w1 * (xb[N + j] - cy) + w2 * (xb[2 * N + j] - cz) + bb;
    op[k] = f2bf(val);
    s += val; q += (double)val * val;
  }
#pragma unroll
  for (int off = 16; off >= 1; off >>= 1) { s += __shfl_xor(s, off, 32); q += __shfl_xor(q, off, 32); }
  if ((threadIdx.x & 31) == 0) {  // wave: same b, same ch -> same group
    const int g = ch >> 4;
    atomicAdd(&stats[(b * GROUPS + g) * 2 + 0], s);
    atomicAdd(&stats[(b * GROUPS + g) * 2 + 1], q);
  }
}

__global__ void finalize_stats_kernel(const double* __restrict__ in,
                                      float* __restrict__ out, float cnt) {
  const int t = threadIdx.x;
  if (t < 2 * GROUPS) {
    const double m = in[t * 2] / (double)cnt;
    const double v = in[t * 2 + 1] / (double)cnt - m * m;
    out[t * 2] = (float)m;
    out[t * 2 + 1] = rsqrtf((float)v + GEPS);
  }
}

// cb = W_att1 @ b_pos2 + b_att1
__global__ void cb_kernel(const float* __restrict__ Watt1, const float* __restrict__ bpos2,
                          const float* __restrict__ batt1, float* __restrict__ cb) {
  const int t = threadIdx.x;
  float s = batt1[t];
  for (int r = 0; r < FEAT; ++r) s += Watt1[t * FEAT + r] * bpos2[r];
  cb[t] = s;
}

// =====================  att1 = Wc @ leaky(gnorm(pos1)) + q2 - gather(k2) + cb  =====================
// 128 threads, 4 points/block; each wave one point (its 16 neighbors = 16 WMMA cols).
__global__ __launch_bounds__(128) void att1_kernel(
    const bf16* __restrict__ pos1, const float* __restrict__ pstat,
    const float* __restrict__ gpos, const float* __restrict__ bepos,
    const float* __restrict__ Wc, const float* __restrict__ cb,
    const float* __restrict__ q2, const float* __restrict__ k2,
    const int* __restrict__ idx, bf16* __restrict__ att1,
    double* __restrict__ stats, int N) {
  __shared__ bf16 Wl[FEAT * FEAT];
  __shared__ bf16 Xl[64 * XS];
  __shared__ float cbl[FEAT], gl[FEAT], bl[FEAT];
  const int tid = threadIdx.x;
  for (int i = tid; i < FEAT * FEAT; i += 128) Wl[i] = f2bf(Wc[i]);
  if (tid < FEAT) { cbl[tid] = cb[tid]; gl[tid] = gpos[tid]; bl[tid] = bepos[tid]; }
  __syncthreads();
  const int p0 = blockIdx.x * 4;
  for (int e = tid; e < 64 * FEAT; e += 128) {
    const int j = e & 63, ch = e >> 6;
    const int pl = j >> 4, k = j & 15;
    const int p = p0 + pl, b = p / N, n = p - b * N;
    const float x = bf2f(pos1[(((size_t)b * FEAT + ch) * N + n) * KNN + k]);
    const int g = ch >> 4;
    const float mean = pstat[(b * GROUPS + g) * 2], rs = pstat[(b * GROUPS + g) * 2 + 1];
    Xl[j * XS + ch] = f2bf(leaky((x - mean) * rs * gl[ch] + bl[ch]));
  }
  __syncthreads();
  const int wave = tid >> 5, lane = tid & 31, half = lane >> 4, l16 = lane & 15;
  const int p = p0 + wave, b = p / N, n = p - b * N;
  const int jidx = idx[(size_t)p * KNN + l16];
  for (int mt = 0; mt < 8; ++mt) {
    v8f acc;
#pragma unroll
    for (int v = 0; v < 8; ++v) {
      const int ch = mt * 16 + v + 8 * half;
      acc[v] = q2[((size_t)b * FEAT + ch) * N + n] - k2[((size_t)b * FEAT + ch) * N + jidx] + cbl[ch];
    }
#pragma unroll
    for (int kt = 0; kt < 4; ++kt) {
      v16bf a = load_a_frag(Wl, mt, kt, l16, half);
      v16bf bm = load_b_frag(Xl, wave * 16, kt, l16, half);
      acc = __builtin_amdgcn_wmma_f32_16x16x32_bf16(false, a, false, bm, (short)0, acc, false, false);
    }
    double s = 0.0, q = 0.0;
#pragma unroll
    for (int v = 0; v < 8; ++v) {
      const int ch = mt * 16 + v + 8 * half;
      att1[(((size_t)b * FEAT + ch) * N + n) * KNN + l16] = f2bf(acc[v]);
      s += acc[v]; q += (double)acc[v] * acc[v];
    }
#pragma unroll
    for (int off = 16; off >= 1; off >>= 1) { s += __shfl_xor(s, off, 32); q += __shfl_xor(q, off, 32); }
    if (lane == 0) {
      atomicAdd(&stats[(b * GROUPS + mt) * 2 + 0], s);
      atomicAdd(&stats[(b * GROUPS + mt) * 2 + 1], q);
    }
  }
}

// ===== out = softmax_k(W_att2 @ leaky(gnorm(att1)) + b_att2) . gather(v) + new_feat =====
__global__ __launch_bounds__(128) void out_kernel(
    const bf16* __restrict__ att1, const float* __restrict__ astat,
    const float* __restrict__ gatt, const float* __restrict__ beatt,
    const float* __restrict__ Watt2, const float* __restrict__ batt2,
    const float* __restrict__ vfeat, const int* __restrict__ idx,
    const float* __restrict__ nf, float* __restrict__ out, int N) {
  __shared__ bf16 Wl[FEAT * FEAT];
  __shared__ bf16 Xl[64 * XS];
  __shared__ float bl[FEAT], gl[FEAT], el[FEAT];
  const int tid = threadIdx.x;
  for (int i = tid; i < FEAT * FEAT; i += 128) Wl[i] = f2bf(Watt2[i]);
  if (tid < FEAT) { bl[tid] = batt2[tid]; gl[tid] = gatt[tid]; el[tid] = beatt[tid]; }
  __syncthreads();
  const int p0 = blockIdx.x * 4;
  for (int e = tid; e < 64 * FEAT; e += 128) {
    const int j = e & 63, ch = e >> 6;
    const int pl = j >> 4, k = j & 15;
    const int p = p0 + pl, b = p / N, n = p - b * N;
    const float x = bf2f(att1[(((size_t)b * FEAT + ch) * N + n) * KNN + k]);
    const int g = ch >> 4;
    const float mean = astat[(b * GROUPS + g) * 2], rs = astat[(b * GROUPS + g) * 2 + 1];
    Xl[j * XS + ch] = f2bf(leaky((x - mean) * rs * gl[ch] + el[ch]));
  }
  __syncthreads();
  const int wave = tid >> 5, lane = tid & 31, half = lane >> 4, l16 = lane & 15;
  const int p = p0 + wave, b = p / N, n = p - b * N;
  const int jidx = idx[(size_t)p * KNN + l16];
  for (int mt = 0; mt < 8; ++mt) {
    v8f acc;
#pragma unroll
    for (int v = 0; v < 8; ++v) acc[v] = bl[mt * 16 + v + 8 * half];
#pragma unroll
    for (int kt = 0; kt < 4; ++kt) {
      v16bf a = load_a_frag(Wl, mt, kt, l16, half);
      v16bf bm = load_b_frag(Xl, wave * 16, kt, l16, half);
      acc = __builtin_amdgcn_wmma_f32_16x16x32_bf16(false, a, false, bm, (short)0, acc, false, false);
    }
    // softmax over k (= 16-lane groups), then weighted sum of gathered v.
#pragma unroll
    for (int v = 0; v < 8; ++v) {
      const int ch = mt * 16 + v + 8 * half;
      const float x = acc[v];
      float m = x;
      m = fmaxf(m, __shfl_xor(m, 8, 32)); m = fmaxf(m, __shfl_xor(m, 4, 32));
      m = fmaxf(m, __shfl_xor(m, 2, 32)); m = fmaxf(m, __shfl_xor(m, 1, 32));
      const float ex = __expf(x - m);
      float s = ex;
      s += __shfl_xor(s, 8, 32); s += __shfl_xor(s, 4, 32);
      s += __shfl_xor(s, 2, 32); s += __shfl_xor(s, 1, 32);
      float r = (ex / s) * vfeat[((size_t)b * FEAT + ch) * N + jidx];
      r += __shfl_xor(r, 8, 32); r += __shfl_xor(r, 4, 32);
      r += __shfl_xor(r, 2, 32); r += __shfl_xor(r, 1, 32);
      if (l16 == 0) out[((size_t)b * FEAT + ch) * N + n] = r + nf[((size_t)b * FEAT + ch) * N + n];
    }
  }
}

// ===== final: leaky(gnorm(post)) -> d_out =====
__global__ void final_kernel(const float* __restrict__ post, const float* __restrict__ stat,
                             const float* __restrict__ g, const float* __restrict__ be,
                             float* __restrict__ y, int N, int total) {
  const int t = blockIdx.x * 256 + threadIdx.x;
  if (t >= total) return;
  const int b = t / (FEAT * N);
  const int ch = (t - b * FEAT * N) / N;
  const int gi = (b * GROUPS + (ch >> 4)) * 2;
  y[t] = leaky((post[t] - stat[gi]) * stat[gi + 1] * g[ch] + be[ch]);
}

extern "C" void kernel_launch(void* const* d_in, const int* in_sizes, int n_in,
                              void* d_out, int out_size, void* d_ws, size_t ws_size,
                              hipStream_t stream) {
  (void)n_in; (void)out_size; (void)ws_size;
  const float* xyz    = (const float*)d_in[0];
  const float* feat   = (const float*)d_in[1];
  const float* W_pre  = (const float*)d_in[2];
  const float* b_pre  = (const float*)d_in[3];
  const float* W_pos1 = (const float*)d_in[4];
  const float* b_pos1 = (const float*)d_in[5];
  const float* g_pos1 = (const float*)d_in[6];
  const float* be_pos1= (const float*)d_in[7];
  const float* W_pos2 = (const float*)d_in[8];
  const float* b_pos2 = (const float*)d_in[9];
  const float* W_q    = (const float*)d_in[10];
  const float* b_q    = (const float*)d_in[11];
  const float* W_k    = (const float*)d_in[12];
  const float* b_k    = (const float*)d_in[13];
  const float* W_v    = (const float*)d_in[14];
  const float* b_v    = (const float*)d_in[15];
  const float* W_att1 = (const float*)d_in[16];
  const float* b_att1 = (const float*)d_in[17];
  const float* g_att1 = (const float*)d_in[18];
  const float* be_att1= (const float*)d_in[19];
  const float* W_att2 = (const float*)d_in[20];
  const float* b_att2 = (const float*)d_in[21];
  const float* W_post = (const float*)d_in[22];
  const float* b_post = (const float*)d_in[23];
  const float* g_post = (const float*)d_in[24];
  const float* be_post= (const float*)d_in[25];

  const int B = 2;
  const int N = in_sizes[0] / (3 * B);   // 8192
  const int colsBN = B * N;

  char* ws = (char*)d_ws;
  size_t off = 0;
  auto alloc = [&](size_t bytes) -> void* {
    void* p = ws + off;
    off = (off + bytes + 255) & ~(size_t)255;
    return p;
  };
  int*    idx   = (int*)   alloc((size_t)B * N * KNN * sizeof(int));
  float*  nf    = (float*) alloc((size_t)B * FEAT * N * sizeof(float));
  float*  qbuf  = (float*) alloc((size_t)B * FEAT * N * sizeof(float));  // q, later 'out'
  float*  kbuf  = (float*) alloc((size_t)B * FEAT * N * sizeof(float));  // k, later 'post'
  float*  vbuf  = (float*) alloc((size_t)B * FEAT * N * sizeof(float));
  float*  q2    = (float*) alloc((size_t)B * FEAT * N * sizeof(float));
  float*  k2    = (float*) alloc((size_t)B * FEAT * N * sizeof(float));
  float*  Wc    = (float*) alloc((size_t)FEAT * FEAT * sizeof(float));
  float*  cb    = (float*) alloc((size_t)FEAT * sizeof(float));
  double* stats = (double*)alloc((size_t)3 * B * GROUPS * 2 * sizeof(double));
  float*  statF = (float*) alloc((size_t)3 * B * GROUPS * 2 * sizeof(float));
  bf16*   pos1  = (bf16*)  alloc((size_t)B * FEAT * N * KNN * sizeof(bf16));
  bf16*   att1  = (bf16*)  alloc((size_t)B * FEAT * N * KNN * sizeof(bf16));

  double* stPos  = stats;
  double* stAtt  = stats + (size_t)B * GROUPS * 2;
  double* stPost = stats + (size_t)2 * B * GROUPS * 2;
  float*  sfPos  = statF;
  float*  sfAtt  = statF + (size_t)B * GROUPS * 2;
  float*  sfPost = statF + (size_t)2 * B * GROUPS * 2;

  hipMemsetAsync(stats, 0, (size_t)3 * B * GROUPS * 2 * sizeof(double), stream);

  knn_kernel<<<B * N / 256, 256, 0, stream>>>(xyz, idx, N);

  gemm128_kernel<<<colsBN / 64, 128, 0, stream>>>(W_pre, feat, b_pre, nf, N, colsBN, nullptr);
  gemm128_kernel<<<colsBN / 64, 128, 0, stream>>>(W_q, nf, b_q, qbuf, N, colsBN, nullptr);
  gemm128_kernel<<<colsBN / 64, 128, 0, stream>>>(W_k, nf, b_k, kbuf, N, colsBN, nullptr);
  gemm128_kernel<<<colsBN / 64, 128, 0, stream>>>(W_v, nf, b_v, vbuf, N, colsBN, nullptr);
  gemm128_kernel<<<colsBN / 64, 128, 0, stream>>>(W_att1, qbuf, nullptr, q2, N, colsBN, nullptr);
  gemm128_kernel<<<colsBN / 64, 128, 0, stream>>>(W_att1, kbuf, nullptr, k2, N, colsBN, nullptr);
  gemm128_kernel<<<FEAT / 64, 128, 0, stream>>>(W_att1, W_pos2, nullptr, Wc, FEAT, FEAT, nullptr);
  cb_kernel<<<1, FEAT, 0, stream>>>(W_att1, b_pos2, b_att1, cb);

  pos1_kernel<<<(B * FEAT * N) / 256, 256, 0, stream>>>(xyz, idx, W_pos1, b_pos1, pos1, stPos, N);
  finalize_stats_kernel<<<1, 32, 0, stream>>>(stPos, sfPos, (float)((FEAT / GROUPS) * (size_t)N * KNN));

  att1_kernel<<<B * N / 4, 128, 0, stream>>>(pos1, sfPos, g_pos1, be_pos1, Wc, cb,
                                             q2, k2, idx, att1, stAtt, N);
  finalize_stats_kernel<<<1, 32, 0, stream>>>(stAtt, sfAtt, (float)((FEAT / GROUPS) * (size_t)N * KNN));

  // qbuf is dead past q2 -> reuse as 'out'
  out_kernel<<<B * N / 4, 128, 0, stream>>>(att1, sfAtt, g_att1, be_att1, W_att2, b_att2,
                                            vbuf, idx, nf, qbuf, N);

  // kbuf is dead past k2 -> reuse as 'post'; accumulate post-gnorm stats in the GEMM.
  gemm128_kernel<<<colsBN / 64, 128, 0, stream>>>(W_post, qbuf, b_post, kbuf, N, colsBN, stPost);
  finalize_stats_kernel<<<1, 32, 0, stream>>>(stPost, sfPost, (float)((FEAT / GROUPS) * (size_t)N));

  final_kernel<<<(B * FEAT * N + 255) / 256, 256, 0, stream>>>(kbuf, sfPost, g_post, be_post,
                                                               (float*)d_out, N, B * FEAT * N);
}
